// GNNBaseline_8899172237601
// MI455X (gfx1250) — matrix operand order repaired and checked by reading, not compiled, not run on hardware
//
#include <hip/hip_runtime.h>
#include <hip/hip_bf16.h>

typedef _Float16 v16h __attribute__((ext_vector_type(16)));
typedef _Float16 v4h  __attribute__((ext_vector_type(4)));
typedef __fp16   v2fp __attribute__((ext_vector_type(2)));   // cvt_pkrtz result type
typedef float    v8f  __attribute__((ext_vector_type(8)));

#define KDIM 512     // input feature dim
#define HDIM 256     // hidden dim
#define NGRAPH 128
#define GE_TM 32     // block tile rows
#define GE_TK 32     // K chunk
#define LDK  40      // padded K stride in LDS halves (80 B rows, 8B-aligned packs)

// ---------------------------------------------------------------- utilities
__global__ void gcn_zero_f32(float* __restrict__ p, long long n) {
    long long i = (long long)blockIdx.x * blockDim.x + threadIdx.x;
    if (i < n) p[i] = 0.0f;
}

__global__ void gcn_edge_degree(const int* __restrict__ ei, float* __restrict__ deg, int E) {
    int e = blockIdx.x * blockDim.x + threadIdx.x;
    if (e < E) atomicAdd(&deg[ei[E + e]], 1.0f);   // dst row of edge_index
}

__global__ void gcn_dinv(float* __restrict__ deg, int N) {
    int i = blockIdx.x * blockDim.x + threadIdx.x;
    if (i < N) deg[i] = rsqrtf(deg[i] + 1.0f);     // in-place: deg -> dinv
}

// ---------------------------------------------------------------- WMMA GEMM
// H0[N,256] = X[N,512] @ W[512,256]; f32 -> f16 via v_cvt_pk, f32 accumulate.
// Block: 256 threads = 8 waves (2 along M x 4 along N). Block tile 32x256.
__global__ __launch_bounds__(256) void gcn_gemm_wmma(const float* __restrict__ X,
                                                     const float* __restrict__ W,
                                                     float* __restrict__ H0,
                                                     int Nrows) {
    __shared__ _Float16 As[GE_TM][LDK];     // [m][k]
    __shared__ _Float16 Bs[HDIM][LDK];      // transposed: [n][k]

    const int tid  = threadIdx.x;
    const int lane = tid & 31;
    const int wave = tid >> 5;
    const int wm   = wave >> 2;             // 0..1  (16-row strip)
    const int wn   = wave & 3;              // 0..3  (64-col strip)
    const int rowBase = blockIdx.x * GE_TM;

    v8f acc[4] = {};                        // 4 x 16x16 f32 accumulators

    for (int kk = 0; kk < KDIM; kk += GE_TK) {
        // ---- stage A chunk: 32 rows x 32 k; one packed ds_store_b64/thread
        {
            int r  = tid >> 3;              // 0..31
            int c4 = (tid & 7) * 4;         // 0..28
            int gr = rowBase + r;
            float4 v = make_float4(0.f, 0.f, 0.f, 0.f);
            if (gr < Nrows)
                v = *reinterpret_cast<const float4*>(X + (size_t)gr * KDIM + kk + c4);
            v2fp lo = __builtin_amdgcn_cvt_pkrtz(v.x, v.y);
            v2fp hi = __builtin_amdgcn_cvt_pkrtz(v.z, v.w);
            v4h p;
            p[0] = (_Float16)lo[0]; p[1] = (_Float16)lo[1];
            p[2] = (_Float16)hi[0]; p[3] = (_Float16)hi[1];
            *reinterpret_cast<v4h*>(&As[r][c4]) = p;
        }
        // ---- stage B chunk transposed: 4k x 4n micro-tile per thread,
        //      packed 4-half ds_store_b64 per destination row.
        #pragma unroll
        for (int v = 0; v < 2; ++v) {
            int k4 = (tid >> 6) * 4 + v * 16;   // {0..12} or {16..28}
            int n4 = (tid & 63) * 4;            // 0..252
            float4 w0 = *reinterpret_cast<const float4*>(W + (size_t)(kk + k4 + 0) * HDIM + n4);
            float4 w1 = *reinterpret_cast<const float4*>(W + (size_t)(kk + k4 + 1) * HDIM + n4);
            float4 w2 = *reinterpret_cast<const float4*>(W + (size_t)(kk + k4 + 2) * HDIM + n4);
            float4 w3 = *reinterpret_cast<const float4*>(W + (size_t)(kk + k4 + 3) * HDIM + n4);
            const float* f0 = reinterpret_cast<const float*>(&w0);
            const float* f1 = reinterpret_cast<const float*>(&w1);
            const float* f2 = reinterpret_cast<const float*>(&w2);
            const float* f3 = reinterpret_cast<const float*>(&w3);
            #pragma unroll
            for (int dn = 0; dn < 4; ++dn) {
                v2fp a  = __builtin_amdgcn_cvt_pkrtz(f0[dn], f1[dn]);
                v2fp bb = __builtin_amdgcn_cvt_pkrtz(f2[dn], f3[dn]);
                v4h p;
                p[0] = (_Float16)a[0];  p[1] = (_Float16)a[1];
                p[2] = (_Float16)bb[0]; p[3] = (_Float16)bb[1];
                *reinterpret_cast<v4h*>(&Bs[n4 + dn][k4]) = p;
            }
        }
        __syncthreads();

        // ---- A fragment (contiguous k = i + 16*(lane>=16), as B) -> two ds_load_b128
        v16h afrag;
        {
            int r  = (lane & 15) + wm * 16;
            int kh = (lane >> 4) * 16;      // upper half-wave: K += 16
            #pragma unroll
            for (int i = 0; i < 16; ++i) {
                int k = i + kh;
                afrag[i] = As[r][k];
            }
        }
        // ---- 4 B fragments + WMMA (B: N = lane&15, K = i + 16*(lane>=16))
        {
            int nb = lane & 15;
            int kb = (lane >> 4) * 16;
            #pragma unroll
            for (int t = 0; t < 4; ++t) {
                v16h bfrag;
                int ncol = wn * 64 + t * 16 + nb;
                #pragma unroll
                for (int i = 0; i < 16; ++i)
                    bfrag[i] = Bs[ncol][kb + i];
                acc[t] = __builtin_amdgcn_wmma_f32_16x16x32_f16(
                    /*neg_a=*/false, afrag, /*neg_b=*/false, bfrag,
                    /*c_mod=*/(short)0, acc[t], /*reuse_a=*/false, /*reuse_b=*/false);
            }
        }
        __syncthreads();
    }

    // ---- store D (M = e + 8*(lane>=16), N = lane&15); one bound check/row
    {
        int colBase = wn * 64 + (lane & 15);
        #pragma unroll
        for (int e = 0; e < 8; ++e) {
            int row = rowBase + wm * 16 + e + (lane >> 4) * 8;
            if (row < Nrows) {
                float* dst = H0 + (size_t)row * HDIM + colBase;
                dst[0]  = acc[0][e];
                dst[16] = acc[1][e];
                dst[32] = acc[2][e];
                dst[48] = acc[3][e];
            }
        }
    }
}

// -------------------------------------------------- self-loop term + bias
__global__ void gcn_self_bias(const float* __restrict__ h0, const float* __restrict__ dinv,
                              const float* __restrict__ b, float* __restrict__ out,
                              long long total) {
    long long i = (long long)blockIdx.x * blockDim.x + threadIdx.x;
    if (i >= total) return;
    int node = (int)(i >> 8);
    int c    = (int)(i & (HDIM - 1));
    float di = dinv[node];
    out[i] = h0[i] * di * di + b[c];
}

// -------------------------------------------------- warp-per-edge aggregation
__global__ void gcn_aggregate(const float* __restrict__ h0, const int* __restrict__ ei,
                              const float* __restrict__ dinv, float* __restrict__ out,
                              int E) {
    int warp = (int)(((long long)blockIdx.x * blockDim.x + threadIdx.x) >> 5);
    int lane = threadIdx.x & 31;
    if (warp >= E) return;
    int src = ei[warp];
    int dst = ei[E + warp];
    const float* hs = h0 + (size_t)src * HDIM;
    __builtin_prefetch(hs + lane * 4, 0, 0);       // global_prefetch_b8: warm gathered row
    float norm = dinv[src] * dinv[dst];
    float* od = out + (size_t)dst * HDIM;
    #pragma unroll
    for (int c = lane; c < HDIM; c += 32)
        atomicAdd(od + c, hs[c] * norm);
}

// -------------------------------------------------- ReLU + mean-pool (sum + count)
__global__ void gcn_pool(const float* __restrict__ h, const int* __restrict__ batch,
                         float* __restrict__ pooled, float* __restrict__ cnt, int Nrows) {
    int node = blockIdx.x;
    if (node >= Nrows) return;
    int c = threadIdx.x;
    int g = batch[node];
    float v = h[(size_t)node * HDIM + c];
    v = v > 0.0f ? v : 0.0f;
    atomicAdd(&pooled[(size_t)g * HDIM + c], v);
    if (c == 0) atomicAdd(&cnt[g], 1.0f);
}

// -------------------------------------------------- MLP head: [G,256]->[G,16]->[G,10]
__global__ void gcn_mlp(const float* __restrict__ pooled, const float* __restrict__ cnt,
                        const float* __restrict__ W1, const float* __restrict__ b1,
                        const float* __restrict__ W2, const float* __restrict__ b2,
                        float* __restrict__ out) {
    int g = threadIdx.x;
    if (g >= NGRAPH) return;
    float inv = 1.0f / fmaxf(cnt[g], 1.0f);
    const float* pg = pooled + (size_t)g * HDIM;
    float z[16];
    #pragma unroll
    for (int j = 0; j < 16; ++j) {
        float s = b1[j];
        for (int c = 0; c < HDIM; ++c)
            s += pg[c] * inv * W1[c * 16 + j];
        z[j] = s > 0.0f ? s : 0.0f;
    }
    #pragma unroll
    for (int o = 0; o < 10; ++o) {
        float s = b2[o];
        #pragma unroll
        for (int j = 0; j < 16; ++j)
            s += z[j] * W2[j * 10 + o];
        out[g * 10 + o] = s;
    }
}

// ================================================================ launch
extern "C" void kernel_launch(void* const* d_in, const int* in_sizes, int n_in,
                              void* d_out, int out_size, void* d_ws, size_t ws_size,
                              hipStream_t stream) {
    const float* x   = (const float*)d_in[0];
    const int*   ei  = (const int*)d_in[1];
    const int*   bat = (const int*)d_in[2];
    const float* W   = (const float*)d_in[3];
    const float* b   = (const float*)d_in[4];
    const float* W1  = (const float*)d_in[5];
    const float* b1  = (const float*)d_in[6];
    const float* W2  = (const float*)d_in[7];
    const float* b2  = (const float*)d_in[8];
    float* out = (float*)d_out;

    const int N = in_sizes[0] / KDIM;       // 50000
    const int E = in_sizes[1] / 2;          // 800000

    // workspace layout (floats)
    float* ws     = (float*)d_ws;
    float* h0     = ws;                                   // N*256
    float* hagg   = h0 + (size_t)N * HDIM;                // N*256
    float* deg    = hagg + (size_t)N * HDIM;              // N (becomes dinv)
    float* pooled = deg + N;                              // 128*256
    float* cnt    = pooled + (size_t)NGRAPH * HDIM;       // 128

    const long long zeroN = (long long)N + (long long)NGRAPH * HDIM + NGRAPH;
    gcn_zero_f32<<<(int)((zeroN + 255) / 256), 256, 0, stream>>>(deg, zeroN);

    gcn_edge_degree<<<(E + 255) / 256, 256, 0, stream>>>(ei, deg, E);
    gcn_dinv<<<(N + 255) / 256, 256, 0, stream>>>(deg, N);

    gcn_gemm_wmma<<<(N + GE_TM - 1) / GE_TM, 256, 0, stream>>>(x, W, h0, N);

    const long long total = (long long)N * HDIM;
    gcn_self_bias<<<(int)((total + 255) / 256), 256, 0, stream>>>(h0, deg, b, hagg, total);

    const long long aggThreads = (long long)E * 32;
    gcn_aggregate<<<(int)((aggThreads + 255) / 256), 256, 0, stream>>>(h0, ei, deg, hagg, E);

    gcn_pool<<<N, HDIM, 0, stream>>>(hagg, bat, pooled, cnt, N);

    gcn_mlp<<<1, NGRAPH, 0, stream>>>(pooled, cnt, W1, b1, W2, b2, out);
}
